// PointCloudAE_44641890074844
// MI455X (gfx1250) — matrix operand order repaired and checked by reading, not compiled, not run on hardware
//
#include <hip/hip_runtime.h>
#include <hip/hip_bf16.h>
#include <cstdint>
#include <cstddef>

#define DEVINL __device__ __forceinline__

typedef __attribute__((ext_vector_type(16))) __bf16 bf16x16;
typedef __attribute__((ext_vector_type(8)))  float  f32x8;

// ---- problem constants (from reference) ----
constexpr int   Bn = 8, Nn = 16384;
constexpr int   K0 = 32, K1 = 32, K2n = 16;
constexpr int   M0 = 512, M1 = 16;
constexpr float R0 = 0.1f, R1 = 0.4f, R2 = 1.0f;

// ---- output layout (floats), flat in return order ----
constexpr size_t O_PTS  = 0;                               // (8,16384,3)
constexpr size_t O_C0   = O_PTS  + (size_t)Bn*Nn*3;        // (8,512,3)
constexpr size_t O_C1   = O_C0   + (size_t)Bn*M0*3;        // (8,16,3)
constexpr size_t O_BN   = O_C1   + (size_t)Bn*M1*3;        // (8,16384)
constexpr size_t O_BM0  = O_BN   + (size_t)Bn*Nn;          // (8,512)
constexpr size_t O_BM1  = O_BM0  + (size_t)Bn*M0;          // (8,16)
constexpr size_t O_P0   = O_BM1  + (size_t)Bn*M1;          // (8,16,3)
constexpr size_t O_P1   = O_P0   + (size_t)Bn*K2n*3;       // (8,512,3)
constexpr size_t O_P2   = O_P1   + (size_t)Bn*K2n*K1*3;    // (8,16384,3)
constexpr size_t O_BK2  = O_P2   + (size_t)Bn*K2n*K1*K0*3; // (8,16)
constexpr size_t O_BK21 = O_BK2  + (size_t)Bn*K2n;         // (8,512)
constexpr size_t O_BK210= O_BK21 + (size_t)Bn*K2n*K1;      // (8,16384)

DEVINL __bf16 f2bf(float f) {
  unsigned u = __builtin_bit_cast(unsigned, f);
  unsigned r = u + 0x7FFFu + ((u >> 16) & 1u);
  unsigned short h = (unsigned short)(r >> 16);
  return __builtin_bit_cast(__bf16, h);
}
DEVINL __bf16 bf_zero() { return __builtin_bit_cast(__bf16, (unsigned short)0); }

// ---------- WMMA tile loaders (CDNA5 bf16 16x16x32 layouts) ----------
// A 16x32 (row-major source): lanes 0-15 row M=lane, elems 0..7 = K 0..7,
// elems 8..15 = K 16..23; lanes 16-31 row M=lane-16, elems 0..7 = K 8..15,
// elems 8..15 = K 24..31.  Both 8-element runs are contiguous -> b128 loads.
DEVINL bf16x16 load_a_tile(const __bf16* base, int ldk, int mt, int kt, int lane, int mrows) {
  int row = mt * 16 + (lane & 15);
  bf16x16 a;
  if (row < mrows) {
    const __bf16* p = base + (size_t)row * ldk + kt * 32 + ((lane < 16) ? 0 : 8);
#pragma unroll
    for (int i = 0; i < 8; ++i) a[i] = p[i];
#pragma unroll
    for (int i = 0; i < 8; ++i) a[8 + i] = p[16 + i];
  } else {
#pragma unroll
    for (int i = 0; i < 16; ++i) a[i] = bf_zero();
  }
  return a;
}
// B tiles are PRE-PACKED into wave layout: tile t = nt*ktiles + kt holds
// 32 lanes x 16 contiguous bf16 (32B/lane) -> two global_load_b128 per tile.
DEVINL bf16x16 load_b_packed(const __bf16* Wp, int ktiles, int nt, int kt, int lane) {
  const __bf16* p = Wp + (((size_t)nt * ktiles + kt) * 32 + lane) * 16;
  bf16x16 b;
#pragma unroll
  for (int i = 0; i < 16; ++i) b[i] = p[i];
  return b;
}
DEVINL f32x8 wmma_bf16(bf16x16 a, bf16x16 b, f32x8 c) {
  return __builtin_amdgcn_wmma_f32_16x16x32_bf16(false, a, false, b, (short)0, c, false, false);
}

// ---------- pack f32 weight W(Ksrc x Nsrc) into B-tile wave layout ----------
// dst has ktiles*ntiles tiles of 512 bf16; rows K padded to ktiles*32,
// cols padded to ntiles*16, padding = 0.
__global__ void pack_b_kernel(const float* __restrict__ src, __bf16* __restrict__ dst,
                              int Ksrc, int Nsrc, int ktiles, int ntiles) {
  size_t e = (size_t)blockIdx.x * 256 + threadIdx.x;
  size_t tot = (size_t)ktiles * ntiles * 512;
  if (e >= tot) return;
  int t = (int)(e >> 9);
  int r = (int)(e & 511);
  int lane = r >> 4, i = r & 15;
  int nt = t / ktiles, kt = t % ktiles;
  int col = nt * 16 + (lane & 15);
  int k   = kt * 32 + ((lane < 16) ? 0 : 8) + ((i < 8) ? i : 8 + i);
  float v = (k < Ksrc && col < Nsrc) ? src[(size_t)k * Nsrc + col] : 0.0f;
  dst[e] = f2bf(v);
}

// ---------- plain f32 -> bf16 convert (for activations) ----------
__global__ void convert_kernel(const float* __restrict__ src, __bf16* __restrict__ dst, size_t n) {
  size_t e = (size_t)blockIdx.x * 256 + threadIdx.x;
  if (e < n) dst[e] = f2bf(src[e]);
}

// ---------- farthest point sampling: one block (8 waves) per batch ----------
template <int NPER>
__global__ __launch_bounds__(256)
void fps_kernel(const float* __restrict__ pts, int n, int m, float* __restrict__ ctrs) {
  int b = blockIdx.x;
  pts  += (size_t)b * n * 3;
  ctrs += (size_t)b * m * 3;
  __shared__ float s_val[256];
  __shared__ int   s_idx[256];
  __shared__ float s_last[3];
  int t = threadIdx.x;
  float mind[NPER];
#pragma unroll
  for (int i = 0; i < NPER; ++i) mind[i] = 1e10f;
  if (t == 0) {
    s_last[0] = pts[0]; s_last[1] = pts[1]; s_last[2] = pts[2];
    ctrs[0] = pts[0]; ctrs[1] = pts[1]; ctrs[2] = pts[2];
  }
  __syncthreads();
  for (int s = 1; s < m; ++s) {
    float lx = s_last[0], ly = s_last[1], lz = s_last[2];
    // keep this thread's slice of the point cloud warm in the WGP cache
    __builtin_prefetch(pts + 3 * t, 0, 3);
    float best = -1.0f; int bi = 0;
#pragma unroll
    for (int i = 0; i < NPER; ++i) {
      int p = t + (i << 8);
      if (p < n) {
        float dx = pts[p*3] - lx, dy = pts[p*3+1] - ly, dz = pts[p*3+2] - lz;
        float d  = dx*dx + dy*dy + dz*dz;
        float mn = fminf(mind[i], d);
        mind[i] = mn;
        if (mn > best) { best = mn; bi = p; }
      }
    }
    s_val[t] = best; s_idx[t] = bi;
    __syncthreads();
    for (int off = 128; off > 0; off >>= 1) {
      if (t < off && s_val[t + off] > s_val[t]) { s_val[t] = s_val[t+off]; s_idx[t] = s_idx[t+off]; }
      __syncthreads();
    }
    if (t == 0) {
      int bp = s_idx[0];
      float x = pts[bp*3], y = pts[bp*3+1], z = pts[bp*3+2];
      s_last[0] = x; s_last[1] = y; s_last[2] = z;
      ctrs[s*3] = x; ctrs[s*3+1] = y; ctrs[s*3+2] = z;
    }
    __syncthreads();
  }
}

// ---------- ball query: one wave per center, ballot slot assignment ----------
__global__ __launch_bounds__(32)
void ball_kernel(const float* __restrict__ pts, const float* __restrict__ ctrs,
                 int n, int m, int k, float rr, int* __restrict__ idx) {
  int g = blockIdx.x, b = g / m;
  pts += (size_t)b * n * 3;
  const float* ct = ctrs + (size_t)g * 3;
  int* out = idx + (size_t)g * k;
  float cx = ct[0], cy = ct[1], cz = ct[2];
  int lane = threadIdx.x;
  int cnt = 0, first = -1;
  for (int base = 0; base < n && cnt < k; base += 32) {
    int p = base + lane;
    bool valid = false;
    if (p < n) {
      float dx = pts[p*3] - cx, dy = pts[p*3+1] - cy, dz = pts[p*3+2] - cz;
      valid = (dx*dx + dy*dy + dz*dz) <= rr;
    }
    unsigned mask = (unsigned)__ballot(valid);
    if (first < 0 && mask) first = base + __ffs(mask) - 1;
    int pre = __popc(mask & ((1u << lane) - 1u));
    if (valid && (cnt + pre) < k) out[cnt + pre] = p;
    cnt += __popc(mask);
  }
  cnt = (cnt < k) ? cnt : k;
  int f = (first >= 0) ? first : 0;
  for (int j = cnt + lane; j < k; j += 32) out[j] = f;
}

// ---------- gather + relative-coordinate builders (emit bf16 X matrices) ----------
__global__ void build_x0_kernel(const float* __restrict__ pts, const float* __restrict__ c0,
                                const int* __restrict__ idx0, __bf16* __restrict__ X) {
  size_t e = (size_t)blockIdx.x * 256 + threadIdx.x;   // rows0*32 entries
  size_t tot = (size_t)Bn * M0 * K0 * 32;
  if (e >= tot) return;
  int col = (int)(e & 31);
  size_t row = e >> 5;
  int j = (int)(row % K0);
  size_t g = row / K0;           // b*M0 + c
  int b = (int)(g / M0);
  float v = 0.0f;
  if (col < 3) {
    int p = idx0[g * K0 + j];
    v = (pts[((size_t)b * Nn + p) * 3 + col] - c0[g * 3 + col]) * (1.0f / R0);
  }
  X[e] = f2bf(v);
}

__global__ void build_x1_kernel(const float* __restrict__ c0, const float* __restrict__ c1,
                                const float* __restrict__ f0, const int* __restrict__ idx1,
                                __bf16* __restrict__ X) {
  size_t e = (size_t)blockIdx.x * 256 + threadIdx.x;   // rows1*96 entries
  size_t tot = (size_t)Bn * M1 * K1 * 96;
  if (e >= tot) return;
  int col = (int)(e % 96);
  size_t row = e / 96;
  int j = (int)(row % K1);
  size_t g = row / K1;           // b*M1 + ci
  int b = (int)(g / M1);
  float v = 0.0f;
  int p = idx1[g * K1 + j];
  if (col < 3)       v = (c0[((size_t)b * M0 + p) * 3 + col] - c1[g * 3 + col]) * (1.0f / R1);
  else if (col < 67) v = f0[((size_t)b * M0 + p) * 64 + (col - 3)];
  X[e] = f2bf(v);
}

__global__ void build_x2_kernel(const float* __restrict__ c1, const float* __restrict__ f1,
                                __bf16* __restrict__ X) {
  size_t e = (size_t)blockIdx.x * 256 + threadIdx.x;   // 128*160 entries
  size_t tot = (size_t)Bn * M1 * 160;
  if (e >= tot) return;
  int col = (int)(e % 160);
  size_t row = e / 160;          // b*16 + j
  float v = 0.0f;
  if (col < 3)        v = c1[row * 3 + col] * (1.0f / R2);
  else if (col < 131) v = f1[row * 128 + (col - 3)];
  X[e] = f2bf(v);
}

// ---------- fused 2-layer MLP + group max-pool (one wave per group) ----------
template <int G, int KP, int H, int DOUT>
__global__ __launch_bounds__(32)
void enc_mlp_kernel(const __bf16* __restrict__ X, const __bf16* __restrict__ W1p,
                    const float* __restrict__ b1, const __bf16* __restrict__ W2p,
                    const float* __restrict__ b2, float* __restrict__ out) {
  int grp = blockIdx.x, lane = threadIdx.x;
  const __bf16* Xg = X + (size_t)grp * G * KP;
  __shared__ __bf16 Hs[G * H];
  // stage 1: H = relu(X @ W1 + b1), stored bf16 row-major in LDS
  for (int mt = 0; mt < G / 16; ++mt)
    for (int nt = 0; nt < H / 16; ++nt) {
      f32x8 acc = {};
#pragma unroll
      for (int kt = 0; kt < KP / 32; ++kt) {
        bf16x16 a  = load_a_tile(Xg, KP, mt, kt, lane, G);
        bf16x16 bm = load_b_packed(W1p, KP / 32, nt, kt, lane);
        acc = wmma_bf16(a, bm, acc);
      }
      int colB  = nt * 16 + (lane & 15);
      float bia = b1[colB];
      int rbase = mt * 16 + ((lane < 16) ? 0 : 8);
#pragma unroll
      for (int i = 0; i < 8; ++i) {
        float v = acc[i] + bia;
        Hs[(size_t)(rbase + i) * H + colB] = f2bf(v > 0.0f ? v : 0.0f);
      }
    }
  __syncthreads();
  // stage 2: F = H @ W2 + b2, max-pooled over the G rows of the group
  for (int nt = 0; nt < DOUT / 16; ++nt) {
    int colB  = nt * 16 + (lane & 15);
    float bia = b2[colB];
    float part = -1e30f;
    for (int mt = 0; mt < G / 16; ++mt) {
      f32x8 acc = {};
#pragma unroll
      for (int kt = 0; kt < H / 32; ++kt) {
        bf16x16 a  = load_a_tile(Hs, H, mt, kt, lane, G);
        bf16x16 bm = load_b_packed(W2p, H / 32, nt, kt, lane);
        acc = wmma_bf16(a, bm, acc);
      }
#pragma unroll
      for (int i = 0; i < 8; ++i) part = fmaxf(part, acc[i] + bia);
    }
    // lane c holds max of rows 0..7 (of each M-tile), lane c+16 rows 8..15
    part = fmaxf(part, __shfl_xor(part, 16, 32));
    if (lane < 16) out[(size_t)grp * DOUT + colB] = part;
  }
}

// ---------- decoder GEMM: C = A(MxK) @ W(KxN) + bias, 4 N-tiles per wave ----------
__global__ __launch_bounds__(32)
void gemm_bias4_kernel(const __bf16* __restrict__ A, const __bf16* __restrict__ Wp,
                       const float* __restrict__ bias, float* __restrict__ C,
                       int Mr, int K, int Ncol, int ktiles) {
  int mt = blockIdx.x, ntg = blockIdx.y, lane = threadIdx.x;
  f32x8 acc[4] = {};
  for (int kt = 0; kt < ktiles; ++kt) {
    bf16x16 a = load_a_tile(A, K, mt, kt, lane, Mr);   // A loaded once per k-step
#pragma unroll
    for (int q = 0; q < 4; ++q) {                      // feeds 4 WMMAs
      bf16x16 bm = load_b_packed(Wp, ktiles, ntg * 4 + q, kt, lane);
      acc[q] = wmma_bf16(a, bm, acc[q]);
    }
  }
  int rbase = mt * 16 + ((lane < 16) ? 0 : 8);
#pragma unroll
  for (int q = 0; q < 4; ++q) {
    int col = (ntg * 4 + q) * 16 + (lane & 15);
    if (col < Ncol) {
      float bv = bias[col];
#pragma unroll
      for (int i = 0; i < 8; ++i) {
        int r = rbase + i;
        if (r < Mr) C[(size_t)r * Ncol + col] = acc[q][i] + bv;
      }
    }
  }
}

// ---------- decoder post-processing / composition ----------
__global__ void dec0_post_kernel(const float* __restrict__ d0, float* __restrict__ out,
                                 __bf16* __restrict__ g0) {
  size_t e = (size_t)blockIdx.x * 256 + threadIdx.x;   // 8*2096
  size_t tot = (size_t)Bn * 2096;
  if (e >= tot) return;
  int b = (int)(e / 2096), col = (int)(e % 2096);
  int j = col / 131, c = col % 131;
  float v = d0[e];
  if (c < 3) out[O_P0 + ((size_t)(b * K2n + j) * 3 + c)] = v * R2;
  else       g0[(size_t)(b * K2n + j) * 128 + (c - 3)] = f2bf(v);
}

__global__ void dec1_post_kernel(const float* __restrict__ d1, float* __restrict__ out,
                                 __bf16* __restrict__ g1) {
  size_t e = (size_t)blockIdx.x * 256 + threadIdx.x;   // 128*2144
  size_t tot = (size_t)Bn * K2n * 2144;
  if (e >= tot) return;
  size_t row = e / 2144;  int col = (int)(e % 2144);
  int b = (int)(row / K2n), j = (int)(row % K2n);
  int i = col / 67, c = col % 67;
  int u = j * K1 + i;
  float v = d1[e];
  if (c < 3)
    out[O_P1 + ((size_t)(b * 512 + u) * 3 + c)] =
        out[O_P0 + ((size_t)(b * K2n + j) * 3 + c)] + v * R1;
  else
    g1[((size_t)b * 512 + u) * 64 + (c - 3)] = f2bf(v);
}

__global__ void dec2_post_kernel(const float* __restrict__ d2, float* __restrict__ out) {
  size_t e = (size_t)blockIdx.x * 256 + threadIdx.x;   // 4096*96
  size_t tot = (size_t)Bn * 512 * 96;
  if (e >= tot) return;
  size_t row = e / 96;  int col = (int)(e % 96);
  int b = (int)(row / 512), u = (int)(row % 512);
  int i2 = col / 3, c = col % 3;
  float v = d2[e];
  out[O_P2 + ((size_t)b * 16384 + (size_t)u * 32 + i2) * 3 + c] =
      out[O_P1 + ((size_t)(b * 512 + u) * 3 + c)] + v * R0;
}

// ---------- trivial copies / batch-vector fills ----------
__global__ void copy_f32_kernel(const float* __restrict__ s, float* __restrict__ d, size_t n) {
  size_t i = (size_t)blockIdx.x * 256 + threadIdx.x;
  if (i < n) d[i] = s[i];
}
__global__ void fill_batch_kernel(float* __restrict__ d, int per, size_t total) {
  size_t i = (size_t)blockIdx.x * 256 + threadIdx.x;
  if (i < total) d[i] = (float)(i / per);
}

static inline int gblk(size_t n) { return (int)((n + 255) / 256); }
static inline size_t packed_elems(int ktiles, int ntiles) { return (size_t)ktiles * ntiles * 512; }

extern "C" void kernel_launch(void* const* d_in, const int* in_sizes, int n_in,
                              void* d_out, int out_size, void* d_ws, size_t ws_size,
                              hipStream_t stream) {
  (void)in_sizes; (void)n_in; (void)out_size; (void)ws_size;
  const float* points = (const float*)d_in[0];
  const float* e0w1 = (const float*)d_in[1];  const float* e0b1 = (const float*)d_in[2];
  const float* e0w2 = (const float*)d_in[3];  const float* e0b2 = (const float*)d_in[4];
  const float* e1w1 = (const float*)d_in[5];  const float* e1b1 = (const float*)d_in[6];
  const float* e1w2 = (const float*)d_in[7];  const float* e1b2 = (const float*)d_in[8];
  const float* e2w1 = (const float*)d_in[9];  const float* e2b1 = (const float*)d_in[10];
  const float* e2w2 = (const float*)d_in[11]; const float* e2b2 = (const float*)d_in[12];
  const float* dc0w = (const float*)d_in[13]; const float* dc0b = (const float*)d_in[14];
  const float* dc1w = (const float*)d_in[15]; const float* dc1b = (const float*)d_in[16];
  const float* dc2w = (const float*)d_in[17]; const float* dc2b = (const float*)d_in[18];
  float* out = (float*)d_out;

  // packed-weight tile grids: (ktiles = Kpad/32, ntiles = Npad/16)
  // decoders padded to multiples of 4 n-tiles for the 4-wide GEMM.
  constexpr int KT_E0W1 = 1,  NT_E0W1 = 4;    // 3x64    -> 32x64
  constexpr int KT_E0W2 = 2,  NT_E0W2 = 4;    // 64x64
  constexpr int KT_E1W1 = 3,  NT_E1W1 = 8;    // 67x128  -> 96x128
  constexpr int KT_E1W2 = 4,  NT_E1W2 = 8;    // 128x128
  constexpr int KT_E2W1 = 5,  NT_E2W1 = 16;   // 131x256 -> 160x256
  constexpr int KT_E2W2 = 8,  NT_E2W2 = 16;   // 256x256
  constexpr int KT_D0   = 8,  NT_D0   = 132;  // 256x2096 -> 132 tiles (33 groups)
  constexpr int KT_D1   = 4,  NT_D1   = 136;  // 128x2144 -> 136 tiles (34 groups)
  constexpr int KT_D2   = 2,  NT_D2   = 8;    // 64x96    ->   8 tiles (2 groups)

  // ---- carve workspace ----
  char* w = (char*)d_ws;
  size_t off = 0;
  auto alloc = [&](size_t bytes) -> void* {
    off = (off + 255) & ~(size_t)255;
    void* p = w + off; off += bytes; return p;
  };
  float*  c0f  = (float*) alloc((size_t)Bn*M0*3*4);
  float*  c1f  = (float*) alloc((size_t)Bn*M1*3*4);
  int*    idx0 = (int*)   alloc((size_t)Bn*M0*K0*4);
  int*    idx1 = (int*)   alloc((size_t)Bn*M1*K1*4);
  __bf16* X0   = (__bf16*)alloc((size_t)Bn*M0*K0*32*2);
  float*  f0   = (float*) alloc((size_t)Bn*M0*64*4);
  __bf16* X1   = (__bf16*)alloc((size_t)Bn*M1*K1*96*2);
  float*  f1   = (float*) alloc((size_t)Bn*M1*128*4);
  __bf16* X2   = (__bf16*)alloc((size_t)Bn*M1*160*2);
  float*  enc  = (float*) alloc((size_t)Bn*256*4);
  __bf16* encb = (__bf16*)alloc((size_t)Bn*256*2);
  __bf16* we0w1= (__bf16*)alloc(packed_elems(KT_E0W1, NT_E0W1)*2);
  __bf16* we0w2= (__bf16*)alloc(packed_elems(KT_E0W2, NT_E0W2)*2);
  __bf16* we1w1= (__bf16*)alloc(packed_elems(KT_E1W1, NT_E1W1)*2);
  __bf16* we1w2= (__bf16*)alloc(packed_elems(KT_E1W2, NT_E1W2)*2);
  __bf16* we2w1= (__bf16*)alloc(packed_elems(KT_E2W1, NT_E2W1)*2);
  __bf16* we2w2= (__bf16*)alloc(packed_elems(KT_E2W2, NT_E2W2)*2);
  __bf16* wd0  = (__bf16*)alloc(packed_elems(KT_D0,   NT_D0  )*2);
  __bf16* wd1  = (__bf16*)alloc(packed_elems(KT_D1,   NT_D1  )*2);
  __bf16* wd2  = (__bf16*)alloc(packed_elems(KT_D2,   NT_D2  )*2);
  float*  d0   = (float*) alloc((size_t)Bn*2096*4);
  __bf16* g0   = (__bf16*)alloc((size_t)Bn*K2n*128*2);
  float*  d1   = (float*) alloc((size_t)Bn*K2n*2144*4);
  __bf16* g1   = (__bf16*)alloc((size_t)Bn*512*64*2);
  float*  d2   = (float*) alloc((size_t)Bn*512*96*4);

  // ---- weights -> packed bf16 B tiles ----
  pack_b_kernel<<<gblk(packed_elems(KT_E0W1,NT_E0W1)), 256, 0, stream>>>(e0w1, we0w1,   3,   64, KT_E0W1, NT_E0W1);
  pack_b_kernel<<<gblk(packed_elems(KT_E0W2,NT_E0W2)), 256, 0, stream>>>(e0w2, we0w2,  64,   64, KT_E0W2, NT_E0W2);
  pack_b_kernel<<<gblk(packed_elems(KT_E1W1,NT_E1W1)), 256, 0, stream>>>(e1w1, we1w1,  67,  128, KT_E1W1, NT_E1W1);
  pack_b_kernel<<<gblk(packed_elems(KT_E1W2,NT_E1W2)), 256, 0, stream>>>(e1w2, we1w2, 128,  128, KT_E1W2, NT_E1W2);
  pack_b_kernel<<<gblk(packed_elems(KT_E2W1,NT_E2W1)), 256, 0, stream>>>(e2w1, we2w1, 131,  256, KT_E2W1, NT_E2W1);
  pack_b_kernel<<<gblk(packed_elems(KT_E2W2,NT_E2W2)), 256, 0, stream>>>(e2w2, we2w2, 256,  256, KT_E2W2, NT_E2W2);
  pack_b_kernel<<<gblk(packed_elems(KT_D0,  NT_D0  )), 256, 0, stream>>>(dc0w, wd0,   256, 2096, KT_D0,   NT_D0);
  pack_b_kernel<<<gblk(packed_elems(KT_D1,  NT_D1  )), 256, 0, stream>>>(dc1w, wd1,   128, 2144, KT_D1,   NT_D1);
  pack_b_kernel<<<gblk(packed_elems(KT_D2,  NT_D2  )), 256, 0, stream>>>(dc2w, wd2,    64,   96, KT_D2,   NT_D2);

  // ---- sampling + grouping ----
  fps_kernel<64><<<Bn, 256, 0, stream>>>(points, Nn, M0, c0f);
  fps_kernel<2> <<<Bn, 256, 0, stream>>>(c0f,    M0, M1, c1f);
  ball_kernel<<<Bn*M0, 32, 0, stream>>>(points, c0f, Nn, M0, K0, R0*R0, idx0);
  ball_kernel<<<Bn*M1, 32, 0, stream>>>(c0f,    c1f, M0, M1, K1, R1*R1, idx1);

  // ---- encoder level 0 ----
  build_x0_kernel<<<gblk((size_t)Bn*M0*K0*32), 256, 0, stream>>>(points, c0f, idx0, X0);
  enc_mlp_kernel<32, 32, 64, 64><<<Bn*M0, 32, 0, stream>>>(X0, we0w1, e0b1, we0w2, e0b2, f0);
  // ---- encoder level 1 ----
  build_x1_kernel<<<gblk((size_t)Bn*M1*K1*96), 256, 0, stream>>>(c0f, c1f, f0, idx1, X1);
  enc_mlp_kernel<32, 96, 128, 128><<<Bn*M1, 32, 0, stream>>>(X1, we1w1, e1b1, we1w2, e1b2, f1);
  // ---- encoder level 2 (global) ----
  build_x2_kernel<<<gblk((size_t)Bn*M1*160), 256, 0, stream>>>(c1f, f1, X2);
  enc_mlp_kernel<16, 160, 256, 256><<<Bn, 32, 0, stream>>>(X2, we2w1, e2b1, we2w2, e2b2, enc);

  // ---- decoders ----
  convert_kernel<<<gblk((size_t)Bn*256), 256, 0, stream>>>(enc, encb, (size_t)Bn*256);
  gemm_bias4_kernel<<<dim3(1,   NT_D0/4), 32, 0, stream>>>(encb, wd0, dc0b, d0, Bn,      256, 2096, KT_D0);
  dec0_post_kernel<<<gblk((size_t)Bn*2096), 256, 0, stream>>>(d0, out, g0);
  gemm_bias4_kernel<<<dim3(8,   NT_D1/4), 32, 0, stream>>>(g0,   wd1, dc1b, d1, Bn*K2n,  128, 2144, KT_D1);
  dec1_post_kernel<<<gblk((size_t)Bn*K2n*2144), 256, 0, stream>>>(d1, out, g1);
  gemm_bias4_kernel<<<dim3(256, NT_D2/4), 32, 0, stream>>>(g1,   wd2, dc2b, d2, Bn*512,   64,   96, KT_D2);
  dec2_post_kernel<<<gblk((size_t)Bn*512*96), 256, 0, stream>>>(d2, out);

  // ---- pass-through outputs + batch vectors ----
  copy_f32_kernel<<<gblk((size_t)Bn*Nn*3), 256, 0, stream>>>(points, out + O_PTS, (size_t)Bn*Nn*3);
  copy_f32_kernel<<<gblk((size_t)Bn*M0*3), 256, 0, stream>>>(c0f,    out + O_C0,  (size_t)Bn*M0*3);
  copy_f32_kernel<<<gblk((size_t)Bn*M1*3), 256, 0, stream>>>(c1f,    out + O_C1,  (size_t)Bn*M1*3);
  fill_batch_kernel<<<gblk((size_t)Bn*Nn),  256, 0, stream>>>(out + O_BN,   Nn,  (size_t)Bn*Nn);
  fill_batch_kernel<<<gblk((size_t)Bn*M0),  256, 0, stream>>>(out + O_BM0,  M0,  (size_t)Bn*M0);
  fill_batch_kernel<<<gblk((size_t)Bn*M1),  256, 0, stream>>>(out + O_BM1,  M1,  (size_t)Bn*M1);
  fill_batch_kernel<<<gblk((size_t)Bn*K2n), 256, 0, stream>>>(out + O_BK2,  K2n, (size_t)Bn*K2n);
  fill_batch_kernel<<<gblk((size_t)Bn*512), 256, 0, stream>>>(out + O_BK21, 512, (size_t)Bn*512);
  fill_batch_kernel<<<gblk((size_t)Bn*16384),256,0, stream>>>(out + O_BK210,16384,(size_t)Bn*16384);
}
